// ImprovedSuperPointContrastiveLoss2_7146825581129
// MI455X (gfx1250) — matrix operand order repaired and checked by reading, not compiled, not run on hardware
//
#include <hip/hip_runtime.h>

// ---------------------------------------------------------------------------
// ImprovedSuperPointContrastiveLoss for MI455X (gfx1250, wave32, WMMA).
//
// Bandwidth-bound: stream 102.4 MB of rawPoint_feat once (~4.4 us floor at
// 23.3 TB/s); normalized superpoint table stored once as bf16 (640 KB, L2
// resident). Per-pair dots are computed 16-at-a-time as the diagonal of a
// 16x16x64 matmul via v_wmma_f32_16x16x32_bf16 (2 chained WMMAs per logit
// slot, 11 slots per 16-row tile). 1/||r|| and 1/TEMPERATURE are folded into
// a post-scale of the diagonal so WMMA consumes raw bf16(r).
//
// This revision uses 32-bit unsigned byte offsets from uniform bases for all
// divergent loads so the backend emits global_load ... saddr (GVS) instead of
// per-lane 64-bit address chains (cuts VALU in the gather loop).
//
// Workspace: [0..7] double accumulator, [256 ...] bf16 s_norm (M*64*2 B).
// ---------------------------------------------------------------------------

typedef __attribute__((ext_vector_type(16))) __bf16 v16bf;
typedef __attribute__((ext_vector_type(8)))  float  v8f;

#define FEAT_D   64
#define NUM_NEG  10
#define INV_TEMP (1.0f / 0.07f)

// Load one K-chunk (32 of the 64 K-values) of the A matrix for this lane,
// in contiguous K order, the same order the B loads use:
//   lane half h owns K in {c*32 + h*16 + [0..15]}, element e at K = c*32 + h*16 + e
// rbase is the uniform buffer base; roff1/roff2 are 32-bit byte offsets.
__device__ __forceinline__ v16bf load_a_chunk(const char* __restrict__ rbase,
                                              unsigned roff1, unsigned roff2,
                                              float& ss) {
  float4 q0 = *(const float4*)(rbase + roff1);
  float4 q1 = *(const float4*)(rbase + roff1 + 16);
  float4 q2 = *(const float4*)(rbase + roff2);
  float4 q3 = *(const float4*)(rbase + roff2 + 16);
  float f[16] = {q0.x, q0.y, q0.z, q0.w, q1.x, q1.y, q1.z, q1.w,
                 q2.x, q2.y, q2.z, q2.w, q3.x, q3.y, q3.z, q3.w};
  v16bf a;
#pragma unroll
  for (int e = 0; e < 16; ++e) {
    ss += f[e] * f[e];
    a[e] = (__bf16)f[e];
  }
  return a;
}

__global__ void ispc_init_acc(double* acc) {
  if (threadIdx.x == 0 && blockIdx.x == 0) *acc = 0.0;
}

// L2-normalize superpoint features once, store as bf16 (row-major M x 64).
__global__ void ispc_norm_s(const float* __restrict__ s,
                            __bf16* __restrict__ sbf, int M) {
  int row = blockIdx.x * blockDim.x + threadIdx.x;
  if (row >= M) return;
  const float4* p = (const float4*)(s + (size_t)row * FEAT_D);
  float v[FEAT_D];
  float ss = 0.f;
#pragma unroll
  for (int i = 0; i < 16; ++i) {
    float4 q = p[i];
    v[4 * i + 0] = q.x; v[4 * i + 1] = q.y;
    v[4 * i + 2] = q.z; v[4 * i + 3] = q.w;
    ss += q.x * q.x + q.y * q.y + q.z * q.z + q.w * q.w;
  }
  float scale = 1.0f / fmaxf(sqrtf(ss), 1e-12f);
  v16bf* o = (v16bf*)(sbf + (size_t)row * FEAT_D);
#pragma unroll
  for (int ch = 0; ch < 4; ++ch) {
    v16bf c;
#pragma unroll
    for (int e = 0; e < 16; ++e) c[e] = (__bf16)(v[ch * 16 + e] * scale);
    o[ch] = c;
  }
}

// One wave handles a tile of 16 raw points; 11 logit slots each computed as
// the diagonal of A(16x64) x B(64x16) via two chained bf16 WMMAs.
__global__ void ispc_loss_main(const float* __restrict__ rraw,
                               const int*  __restrict__ posIdx,
                               const int*  __restrict__ negDraws,
                               const __bf16* __restrict__ sbf,
                               double* __restrict__ acc, int N) {
  const int lane  = threadIdx.x & 31;
  const int gwave = blockIdx.x * (blockDim.x >> 5) + (threadIdx.x >> 5);
  const int row0  = gwave * 16;
  if (row0 >= N) return;  // wave-uniform: EXEC stays all-ones for WMMA

  const int h = lane >> 4;   // K-half of this lane
  const int m = lane & 15;   // A-row / B-column owned by this lane
  const int myRow = (row0 + m < N) ? (row0 + m) : (N - 1);

  // --- A tile: uniform base + 32-bit byte offsets (saddr-form loads) ------
  const char* rbase = (const char*)rraw;
  const unsigned rowOff = (unsigned)myRow * (unsigned)(FEAT_D * 4);  // 256 B/row
  const unsigned hOff   = (unsigned)h * 64u;                         // h*16 floats

  float ss = 0.f;
  // chunk 0: K = {h*16..h*16+15}
  v16bf a0 = load_a_chunk(rbase, rowOff + hOff, rowOff + 32u + hOff, ss);
  // chunk 1: same pattern shifted by 32 K (=128 B)
  v16bf a1 = load_a_chunk(rbase, rowOff + 128u + hOff, rowOff + 160u + hOff, ss);
  ss += __shfl_xor(ss, 16);
  const float postScale = (1.0f / fmaxf(sqrtf(ss), 1e-12f)) * INV_TEMP;

  const int pos = posIdx[myRow];

  const char* sbase = (const char*)sbf;
  const unsigned bOff = (unsigned)h * 32u;  // h*16 bf16

  float logits[NUM_NEG + 1];
#pragma unroll
  for (int k = 0; k <= NUM_NEG; ++k) {
    int idx = pos;
    if (k > 0) {
      int dr = negDraws[(size_t)myRow * NUM_NEG + (k - 1)];
      idx = dr + ((dr >= pos) ? 1 : 0);  // skip-positive remap
    }
    // B column n=m: 16 contiguous bf16 per chunk at K = c*32 + h*16.
    // 32-bit byte offset from uniform base -> global_load_b128 saddr form.
    const unsigned soff = (unsigned)idx * (unsigned)(FEAT_D * 2) + bOff;
    v16bf b0 = *(const v16bf*)(sbase + soff);
    v16bf b1 = *(const v16bf*)(sbase + soff + 64u);

    v8f dmat = {};
    dmat = __builtin_amdgcn_wmma_f32_16x16x32_bf16(false, a0, false, b0,
                                                   (short)0, dmat, false, false);
    dmat = __builtin_amdgcn_wmma_f32_16x16x32_bf16(false, a1, false, b1,
                                                   (short)0, dmat, false, false);

    // Diagonal (i,i): lane i (i<8) holds it in VGPR i; lane i+16 (i>=8) in
    // VGPR i-8. Extract with a static select chain (no dynamic vec index).
    int sel = (lane < 8) ? lane : ((lane >= 24) ? (lane - 24) : -1);
    float dv = 0.f;
#pragma unroll
    for (int j = 0; j < 8; ++j)
      if (sel == j) dv = dmat[j];
    logits[k] = dv * postScale;
  }

  // Per-lane logsumexp over the 11 logits (lane owns point row0 + (lane&15))
  float mx = logits[0];
#pragma unroll
  for (int k = 1; k <= NUM_NEG; ++k) mx = fmaxf(mx, logits[k]);
  float se = 0.f;
#pragma unroll
  for (int k = 0; k <= NUM_NEG; ++k) se += __expf(logits[k] - mx);
  float partial = mx + __logf(se) - logits[0];

  const bool diagOwner = (lane < 8) || (lane >= 24);
  const int  point     = row0 + ((lane < 8) ? lane : (lane - 16));
  if (!diagOwner || point >= N) partial = 0.f;

  // wave reduction + one f64 atomic per tile (25k atomics total)
#pragma unroll
  for (int off = 16; off > 0; off >>= 1) partial += __shfl_xor(partial, off);
  if (lane == 0) atomicAdd(acc, (double)partial);
}

__global__ void ispc_finish(const double* acc, float* out, int N) {
  if (threadIdx.x == 0 && blockIdx.x == 0)
    out[0] = (float)(*acc / (double)N);  // LOSS_WEIGHT == 1.0
}

extern "C" void kernel_launch(void* const* d_in, const int* in_sizes, int n_in,
                              void* d_out, int out_size, void* d_ws, size_t ws_size,
                              hipStream_t stream) {
  const float* sfeat    = (const float*)d_in[0];  // (M, 64) f32
  const float* rfeat    = (const float*)d_in[1];  // (N, 64) f32
  const int*   posIdx   = (const int*)d_in[2];    // (N,)    i32
  const int*   negDraws = (const int*)d_in[3];    // (N,10)  i32

  const int M = in_sizes[0] / FEAT_D;
  const int N = in_sizes[1] / FEAT_D;

  double* acc = (double*)d_ws;
  __bf16* sbf = (__bf16*)((char*)d_ws + 256);  // needs 256 + M*64*2 bytes

  ispc_init_acc<<<1, 1, 0, stream>>>(acc);
  ispc_norm_s<<<(M + 127) / 128, 128, 0, stream>>>(sfeat, sbf, M);

  const int tiles  = (N + 15) / 16;          // one wave per 16-row tile
  const int wavesB = 8;                      // 256 threads = 8 waves / block
  const int blocks = (tiles + wavesB - 1) / wavesB;
  ispc_loss_main<<<blocks, 32 * wavesB, 0, stream>>>(rfeat, posIdx, negDraws,
                                                     sbf, acc, N);
  ispc_finish<<<1, 1, 0, stream>>>(acc, (float*)d_out, N);
}
